// CombinedLossI_44830868636253
// MI455X (gfx1250) — matrix operand order repaired and checked by reading, not compiled, not run on hardware
//
#include <hip/hip_runtime.h>

// ---------------------------------------------------------------------------
// Types for CDNA5 WMMA
// ---------------------------------------------------------------------------
typedef __attribute__((ext_vector_type(16))) __bf16 v16bf;
typedef __attribute__((ext_vector_type(8)))  __bf16 v8bf;
typedef __attribute__((ext_vector_type(8)))  float  v8f;

// ---------------------------------------------------------------------------
// Problem constants
// ---------------------------------------------------------------------------
#define Q_DIM     2048
#define B_DIM     256
#define MAX_STEP  50
#define D_EMB     256
#define K_TOTAL   102400          // Q * MAX_STEP
#define KSPLIT    25
#define KCHUNK    (K_TOTAL / KSPLIT)   // 4096
#define LDS_STR   40              // 32 + 8 pad (bf16 elems) to break bank conflicts

// workspace layout (float offsets)
#define GRAM_OFF  0
#define GRAM_N    (9 * 65536)     // 589824
#define KD_OFF    (GRAM_OFF + GRAM_N)       // 589824
#define NCE_OFF   (KD_OFF + 1)              // 589825
#define SUP_OFF   (NCE_OFF + 1)             // 589826
#define PC_OFF    (SUP_OFF + 2)             // 589828
#define NT_SUP    (B_DIM * (MAX_STEP - 1))  // 12544
#define PT_OFF    (PC_OFF + NT_SUP)
#define PE_OFF    (PT_OFF + NT_SUP)
#define AA_OFF    (PE_OFF + NT_SUP)
#define ZERO_N    (SUP_OFF + 1)             // grams + kd + nce + sup

__global__ __launch_bounds__(256) void zero_kernel(float* ws, int n) {
    int i = blockIdx.x * 256 + threadIdx.x;
    if (i < n) ws[i] = 0.0f;
}

// ---------------------------------------------------------------------------
// Gram kernel: G[g] = A * B^T (A,B are 256 x 102400 fp32, scaled by 2 = 1/T_KD)
// block = 128 threads (4 waves); each block computes a 128x64 tile over a K
// chunk; each wave owns 32 output rows (2 A-frags x 4 B-frags = 8 WMMAs/step).
// grid = (KSPLIT, 8 tiles, 9 grams)
// ---------------------------------------------------------------------------
struct GramArgs { const float* src[6]; float* gram; };

__global__ __launch_bounds__(128) void gram_kernel(GramArgs args) {
    const int kblk = blockIdx.x;           // K chunk
    const int tile = blockIdx.y;           // 0..7
    const int g    = blockIdx.z;           // 0..8
    const int pair = g / 3, kind = g % 3;

    const float* xs = args.src[pair];      // student logits
    const float* yt = args.src[3 + pair];  // teacher logits
    const float* X; const float* Y;
    if (kind == 0)      { X = xs; Y = yt; }   // G_xy
    else if (kind == 1) { X = xs; Y = xs; }   // G_xx
    else                { X = yt; Y = yt; }   // G_yy
    float* G = args.gram + (size_t)g * 65536;

    const int bm = (tile & 1) * 128;
    const int bn = (tile >> 1) * 64;

    __shared__ __bf16 ldsX[128 * LDS_STR];
    __shared__ __bf16 ldsY[64 * LDS_STR];

    const int tid  = threadIdx.x;
    const int lane = tid & 31;
    const int wave = tid >> 5;
    const int row  = lane & 15;
    const int grp  = lane >> 4;
    const int m0   = wave * 32;            // 32 rows per wave

    v8f acc[2][4];
#pragma unroll
    for (int m = 0; m < 2; ++m)
#pragma unroll
        for (int n = 0; n < 4; ++n) acc[m][n] = (v8f){0,0,0,0,0,0,0,0};

    const size_t kbase = (size_t)kblk * KCHUNK;
    for (int kk = 0; kk < KCHUNK; kk += 32) {
        const size_t kpos = kbase + kk;
        // cooperative load: X 128x32, Y 64x32 fp32 -> bf16 LDS (x2 = 1/T_KD)
#pragma unroll
        for (int it = 0; it < 8; ++it) {
            int e = it * 128 + tid;        // 0..1023 float4 units
            int r = e >> 3;                // 0..127
            int c = (e & 7) * 4;           // 0..28
            float4 v = *(const float4*)(X + (size_t)(bm + r) * K_TOTAL + kpos + c);
            __bf16* p = &ldsX[r * LDS_STR + c];
            p[0] = (__bf16)(2.0f * v.x); p[1] = (__bf16)(2.0f * v.y);
            p[2] = (__bf16)(2.0f * v.z); p[3] = (__bf16)(2.0f * v.w);
        }
#pragma unroll
        for (int it = 0; it < 4; ++it) {
            int e = it * 128 + tid;        // 0..511 float4 units
            int r = e >> 3;                // 0..63
            int c = (e & 7) * 4;
            float4 v = *(const float4*)(Y + (size_t)(bn + r) * K_TOTAL + kpos + c);
            __bf16* p = &ldsY[r * LDS_STR + c];
            p[0] = (__bf16)(2.0f * v.x); p[1] = (__bf16)(2.0f * v.y);
            p[2] = (__bf16)(2.0f * v.z); p[3] = (__bf16)(2.0f * v.w);
        }
        __syncthreads();

        // A fragments (16x32 each): lane(grp,row) K={grp*8..+7},{16+grp*8..+7}
        const __bf16* ap0 = &ldsX[(m0 + row) * LDS_STR + grp * 8];
        const __bf16* ap1 = ap0 + 16 * LDS_STR;
        v8bf a0lo = *(const v8bf*)ap0;
        v8bf a0hi = *(const v8bf*)(ap0 + 16);
        v8bf a1lo = *(const v8bf*)ap1;
        v8bf a1hi = *(const v8bf*)(ap1 + 16);
        v16bf af0 = __builtin_shufflevector(a0lo, a0hi, 0,1,2,3,4,5,6,7,8,9,10,11,12,13,14,15);
        v16bf af1 = __builtin_shufflevector(a1lo, a1hi, 0,1,2,3,4,5,6,7,8,9,10,11,12,13,14,15);

        // B fragments (32x16): lane(grp,row) col=row, K={grp*8..+7},{16+grp*8..+7}
        v16bf bf[4];
#pragma unroll
        for (int n = 0; n < 4; ++n) {
            const __bf16* bp = &ldsY[(n * 16 + row) * LDS_STR + grp * 8];
            v8bf blo = *(const v8bf*)bp;
            v8bf bhi = *(const v8bf*)(bp + 16);
            bf[n] = __builtin_shufflevector(blo, bhi, 0,1,2,3,4,5,6,7,8,9,10,11,12,13,14,15);
        }

#pragma unroll
        for (int n = 0; n < 4; ++n) {
            acc[0][n] = __builtin_amdgcn_wmma_f32_16x16x32_bf16(
                false, af0, false, bf[n], (short)0, acc[0][n], false, false);
            acc[1][n] = __builtin_amdgcn_wmma_f32_16x16x32_bf16(
                false, af1, false, bf[n], (short)0, acc[1][n], false, false);
        }
        __syncthreads();
    }

    // C layout: lane<16 -> M=v, N=lane ; lane>=16 -> M=8+v, N=lane-16
#pragma unroll
    for (int m = 0; m < 2; ++m) {
#pragma unroll
        for (int n = 0; n < 4; ++n) {
#pragma unroll
            for (int v = 0; v < 8; ++v) {
                int rr = bm + m0 + m * 16 + grp * 8 + v;
                int cc = bn + n * 16 + row;
                atomicAdd(&G[(size_t)rr * 256 + cc], acc[m][n][v]);
            }
        }
    }
}

// ---------------------------------------------------------------------------
// Sinkhorn divergence: 3 blocks (one per pair), 256 threads (one per row)
// ---------------------------------------------------------------------------
__device__ float softmin_row(float eps, const float* __restrict__ G,
                             const float* __restrict__ sqI, const float* __restrict__ sqJ,
                             int i, int transposed, const float* __restrict__ h) {
    const float inv = 1.0f / eps;
    const float si = sqI[i];
    float m = -3.4e38f;
    for (int j = 0; j < 256; ++j) {
        float gv = transposed ? G[(size_t)j * 256 + i] : G[(size_t)i * 256 + j];
        float c  = 0.5f * fmaxf(si + sqJ[j] - 2.0f * gv, 0.0f);
        m = fmaxf(m, h[j] - c * inv);
    }
    float s = 0.0f;
    for (int j = 0; j < 256; ++j) {
        float gv = transposed ? G[(size_t)j * 256 + i] : G[(size_t)i * 256 + j];
        float c  = 0.5f * fmaxf(si + sqJ[j] - 2.0f * gv, 0.0f);
        s += expf(h[j] - c * inv - m);
    }
    return -eps * (m + logf(s));
}

__global__ __launch_bounds__(256) void sinkhorn_kernel(const float* __restrict__ grams,
                                                       float* kd_acc) {
    const int p = blockIdx.x;
    const float* Gxy = grams + (size_t)(p * 3 + 0) * 65536;
    const float* Gxx = grams + (size_t)(p * 3 + 1) * 65536;
    const float* Gyy = grams + (size_t)(p * 3 + 2) * 65536;

    __shared__ float x2[256], y2[256];
    __shared__ float f[256], g[256], fa[256], gb[256];
    __shared__ float ha[256], hb[256], hfa[256], hgb[256];
    __shared__ float red[256];

    const int i = threadIdx.x;
    x2[i] = Gxx[(size_t)i * 257];
    y2[i] = Gyy[(size_t)i * 257];
    f[i] = 0.0f; g[i] = 0.0f; fa[i] = 0.0f; gb[i] = 0.0f;
    __syncthreads();

    const float loga = -logf(256.0f);
    const float logb = loga;
    const float RHO = 250000.0f;                 // _REACH^2
    const float epsl[10] = {1.0f, 1.0f, 0.25f, 0.0625f, 0.015625f, 0.00390625f,
                            9.765625e-4f, 2.44140625e-4f, 6.103515625e-5f, 2.5e-5f};

    for (int e = 0; e < 10; ++e) {
        const float eps = epsl[e];
        const float tau = 1.0f / (1.0f + eps / RHO);
        hb[i]  = logb + g[i]  / eps;
        ha[i]  = loga + f[i]  / eps;
        hfa[i] = loga + fa[i] / eps;
        hgb[i] = logb + gb[i] / eps;
        __syncthreads();
        float ftv = tau * softmin_row(eps, Gxy, x2, y2, i, 0, hb);
        float gtv = tau * softmin_row(eps, Gxy, y2, x2, i, 1, ha);
        float fav = tau * softmin_row(eps, Gxx, x2, x2, i, 0, hfa);
        float gbv = tau * softmin_row(eps, Gyy, y2, y2, i, 0, hgb);
        __syncthreads();
        f[i]  = 0.5f * (f[i]  + ftv);
        g[i]  = 0.5f * (g[i]  + gtv);
        fa[i] = 0.5f * (fa[i] + fav);
        gb[i] = 0.5f * (gb[i] + gbv);
        __syncthreads();
    }

    // final debiasing step (replaces f_ba, g_ab only)
    const float eps = 2.5e-5f;
    const float tau = 1.0f / (1.0f + eps / RHO);
    hb[i] = logb + g[i] / eps;
    ha[i] = loga + f[i] / eps;
    __syncthreads();
    float fb_fin = tau * softmin_row(eps, Gxy, x2, y2, i, 0, hb);
    float g_fin  = tau * softmin_row(eps, Gxy, y2, x2, i, 1, ha);

    const float w  = RHO + eps * 0.5f;
    const float ab = 1.0f / 256.0f;
    float term = ab * (expf(-fa[i] / RHO) - expf(-fb_fin / RHO))
               + ab * (expf(-gb[i] / RHO) - expf(-g_fin  / RHO));

    red[i] = term; __syncthreads();
    for (int s = 128; s > 0; s >>= 1) {
        if (i < s) red[i] += red[i + s];
        __syncthreads();
    }
    if (i == 0) atomicAdd(kd_acc, w * red[0]);
}

// ---------------------------------------------------------------------------
// Supervised loss stage 1: per (b,t) reductions over Q
// ---------------------------------------------------------------------------
__global__ __launch_bounds__(256) void sup_stage1(const float* __restrict__ lc,
                                                  const float* __restrict__ lt,
                                                  const float* __restrict__ le,
                                                  const float* __restrict__ batch,
                                                  float* pcA, float* ptA, float* peA, float* aA) {
    const int t = blockIdx.x;   // 0..48
    const int b = blockIdx.y;   // 0..255
    const int tid = threadIdx.x;

    const float* bb  = batch + ((size_t)b * MAX_STEP + t + 1) * (2 * Q_DIM);
    const float* lcb = lc + ((size_t)b * MAX_STEP + t) * Q_DIM;
    const float* ltb = lt + ((size_t)b * MAX_STEP + t) * Q_DIM;
    const float* leb = le + ((size_t)b * MAX_STEP + t) * Q_DIM;

    float pc = 0.0f, pt = 0.0f, pe = 0.0f, sd = 0.0f;
    for (int q = tid; q < Q_DIM; q += 256) {
        float b1 = bb[q], b2 = bb[Q_DIM + q];
        float d = b1 + b2;
        sd += b1 - b2;
        pc += d / (1.0f + expf(-lcb[q]));
        pt += d / (1.0f + expf(-ltb[q]));
        pe += d / (1.0f + expf(-leb[q]));
    }
    __shared__ float4 red[256];
    red[tid] = make_float4(pc, pt, pe, sd);
    __syncthreads();
    for (int s = 128; s > 0; s >>= 1) {
        if (tid < s) {
            float4 o = red[tid + s]; float4 m = red[tid];
            m.x += o.x; m.y += o.y; m.z += o.z; m.w += o.w;
            red[tid] = m;
        }
        __syncthreads();
    }
    if (tid == 0) {
        float4 r = red[0];
        int idx = b * (MAX_STEP - 1) + t;
        pcA[idx] = r.x; ptA[idx] = r.y; peA[idx] = r.z;
        aA[idx]  = floorf((r.w + 1.0f) * 0.5f);
    }
}

__device__ __forceinline__ float bce(float p, float a) {
    p = fminf(fmaxf(p, 1e-12f), 1.0f - 1e-7f);
    return -(a * logf(p) + (1.0f - a) * log1pf(-p));
}

// stage 2: reverse cummax mask + BCE, one thread per student; single block
__global__ __launch_bounds__(256) void sup_stage2(const float* __restrict__ pcA,
                                                  const float* __restrict__ ptA,
                                                  const float* __restrict__ peA,
                                                  const float* __restrict__ aA,
                                                  float* supOut) {
    const int b = threadIdx.x;
    float run = 0.0f, acc = 0.0f, msum = 0.0f;
    for (int t = MAX_STEP - 2; t >= 0; --t) {
        int idx = b * (MAX_STEP - 1) + t;
        float pc = pcA[idx];
        if (pc > 0.0f) run = 1.0f;
        if (run > 0.0f) {
            float a = aA[idx];
            acc += bce(ptA[idx], a) + bce(pc, a) + bce(peA[idx], a);
            msum += 1.0f;
        }
    }
    float per = acc / fmaxf(msum, 1.0f);
    __shared__ float red[256];
    red[b] = per; __syncthreads();
    for (int s = 128; s > 0; s >>= 1) {
        if (b < s) red[b] += red[b + s];
        __syncthreads();
    }
    if (b == 0) supOut[0] = red[0];
}

// ---------------------------------------------------------------------------
// InfoNCE: one block per (b,t); 7-way dot/norm reduction over D_EMB=256
// ---------------------------------------------------------------------------
__global__ __launch_bounds__(256) void nce_kernel(const float* __restrict__ A,
                                                  const float* __restrict__ P,
                                                  const float* __restrict__ N1,
                                                  const float* __restrict__ N2,
                                                  float* acc) {
    const size_t base = (size_t)blockIdx.x * D_EMB;
    const int tid = threadIdx.x;
    float av = A[base + tid], pv = P[base + tid];
    float n1 = N1[base + tid], n2 = N2[base + tid];

    __shared__ float4 r1[256];
    __shared__ float4 r2[256];
    r1[tid] = make_float4(av * pv, av * n1, av * n2, av * av);
    r2[tid] = make_float4(pv * pv, n1 * n1, n2 * n2, 0.0f);
    __syncthreads();
    for (int s = 128; s > 0; s >>= 1) {
        if (tid < s) {
            float4 o = r1[tid + s]; float4 m = r1[tid];
            m.x += o.x; m.y += o.y; m.z += o.z; m.w += o.w; r1[tid] = m;
            o = r2[tid + s]; m = r2[tid];
            m.x += o.x; m.y += o.y; m.z += o.z; m.w += o.w; r2[tid] = m;
        }
        __syncthreads();
    }
    if (tid == 0) {
        float4 a1 = r1[0], a2 = r2[0];
        float na  = fmaxf(sqrtf(a1.w), 1e-8f);
        float npv = fmaxf(sqrtf(a2.x), 1e-8f);
        float nn1 = fmaxf(sqrtf(a2.y), 1e-8f);
        float nn2 = fmaxf(sqrtf(a2.z), 1e-8f);
        const float invT = 2.0f;  // 1 / T_NCE
        float pos = a1.x / (na * npv) * invT;
        float e1  = a1.y / (na * nn1) * invT;
        float e2  = a1.z / (na * nn2) * invT;
        float m = fmaxf(pos, fmaxf(e1, e2));
        float lse = m + logf(expf(pos - m) + expf(e1 - m) + expf(e2 - m));
        atomicAdd(acc, lse - pos);
    }
}

// ---------------------------------------------------------------------------
// final combine
// ---------------------------------------------------------------------------
__global__ void final_kernel(const float* __restrict__ sup,
                             const float* __restrict__ kd,
                             const float* __restrict__ nce,
                             float* out) {
    out[0] = sup[0] + 0.01f * kd[0] + nce[0] * (1.0f / (B_DIM * MAX_STEP));
}

// ---------------------------------------------------------------------------
// launcher
// ---------------------------------------------------------------------------
extern "C" void kernel_launch(void* const* d_in, const int* in_sizes, int n_in,
                              void* d_out, int out_size, void* d_ws, size_t ws_size,
                              hipStream_t stream) {
    (void)in_sizes; (void)n_in; (void)out_size; (void)ws_size;
    float* ws = (float*)d_ws;

    GramArgs ga;
    for (int i = 0; i < 6; ++i) ga.src[i] = (const float*)d_in[i];
    ga.gram = ws + GRAM_OFF;

    const float* hs = (const float*)d_in[6];   // out_h_student (anchor)
    const float* ht = (const float*)d_in[7];   // out_h_teacher (positive)
    const float* ds = (const float*)d_in[8];   // out_d_student (neg1)
    const float* dt = (const float*)d_in[9];   // out_d_teacher (neg2)
    const float* batch = (const float*)d_in[10];

    // 1. zero accumulators (grams + kd + nce + sup)
    zero_kernel<<<(ZERO_N + 255) / 256, 256, 0, stream>>>(ws, ZERO_N);

    // 2. 9 Gram matrices via bf16 WMMA, K-split across 25 chunks
    dim3 ggrid(KSPLIT, 8, 9);
    gram_kernel<<<ggrid, 128, 0, stream>>>(ga);

    // 3. Sinkhorn divergences (one block per logit pair)
    sinkhorn_kernel<<<3, 256, 0, stream>>>(ws + GRAM_OFF, ws + KD_OFF);

    // 4. supervised loss
    sup_stage1<<<dim3(MAX_STEP - 1, B_DIM), 256, 0, stream>>>(
        (const float*)d_in[0], (const float*)d_in[1], (const float*)d_in[2],
        batch, ws + PC_OFF, ws + PT_OFF, ws + PE_OFF, ws + AA_OFF);
    sup_stage2<<<1, 256, 0, stream>>>(ws + PC_OFF, ws + PT_OFF, ws + PE_OFF,
                                      ws + AA_OFF, ws + SUP_OFF);

    // 5. InfoNCE
    nce_kernel<<<B_DIM * MAX_STEP, 256, 0, stream>>>(hs, ht, ds, dt, ws + NCE_OFF);

    // 6. combine
    final_kernel<<<1, 1, 0, stream>>>(ws + SUP_OFF, ws + KD_OFF, ws + NCE_OFF,
                                      (float*)d_out);
}
